// Net_21818433863930
// MI455X (gfx1250) — matrix operand order repaired and checked by reading, not compiled, not run on hardware
//
#include <hip/hip_runtime.h>
#include <math.h>

typedef __attribute__((ext_vector_type(16))) _Float16 v16h;
typedef __attribute__((ext_vector_type(8)))  _Float16 v8h;
typedef __attribute__((ext_vector_type(8)))  float    v8f;

#define NSAMP 8192
#define NOBJ  8
#define CLS   32
#define NROWS (NSAMP*NOBJ)   // 65536

// ---- fragment bases (units of one fragment = 512 halves = 1024 B) ----
#define FB_WENC  0            // 1x2  frags
#define FB_SW0   2            // +c*2
#define FB_SW1   8            // +c*2
#define FB_RW0   14           // +c*12 (3x4)
#define FB_RW1   50           // +c*4  (2x2)
#define FB_RW2   62           // +c*2
#define FB_AW0   68           // +c*12
#define FB_AW1   104          // +c*4
#define FB_FW0   116          // +c*2
#define FB_FW1   122          // +c*2
#define FB_FW2   128          // +c*2
#define FB_OW0   134          // +c*4
#define FB_OW1   146          // +c*2
#define FB_WAGG1 152          // 3x2
#define FB_WAGG2 158          // 1x2
#define NFRAGS   160

// ---- ws layout (bytes) ----
#define WS_FRAGS   0
#define WS_SF      (NFRAGS*1024)                          // frags = 163840 B
#define WS_CORE    (WS_FRAGS + NFRAGS*1024 + NROWS*CLS*4) // + sF (8 MB)

// ---- core-kernel shared memory carve (bytes). att path runs BEFORE rel
// path so rel2F can alias comb; post buffers alias dead h1. ----
#define SM_COMB   0        // 256*96*2  = 49152  (f16)  dead after rel0
#define SM_REL2F  0        // 256*32*4  = 32768  (f32)  aliases comb
#define SM_H0     49152    // 256*64*2  = 32768  (f16)
#define SM_H1     81920    // 256*32*2  = 16384  (f16)  dead after rel2
#define SM_CATB   81920    // 32*64*2   = 4096   aliases h1
#define SM_PA     86016    // 32*32*2   = 2048   aliases h1
#define SM_PB     88064    // 32*32*2   = 2048   aliases h1
#define SM_SH     98304    // 32*32*2   = 2048
#define SM_SF     100352   // 32*32*4   = 4096
#define SM_ATT2F  104448   // 256*4     = 1024
#define SM_RELSD  105472   // 32*32*4   = 4096
#define SM_SIZE   109568

struct Params {
    const float *z, *Wenc, *benc;
    const float *sw0,*sb0,*sw1,*sb1;
    const float *rw0,*rb0,*rw1,*rb1,*rw2,*rb2;
    const float *aw0,*ab0,*aw1,*ab1,*aw2,*ab2;
    const float *fw0,*fb0,*fw1,*fb1,*fw2,*fb2;
    const float *ow0,*ob0,*ow1,*ob1;
    const float *Wagg1,*bagg1,*Wagg2,*bagg2;
};

// ---------------- WMMA fragment helpers ----------------
// A (16x32 f16): elem e: K = e + (e>=8?8:0) + (lane>=16?8:0), M = lane%16
// => two contiguous 16B LDS loads: halves [kb..kb+7], [kb+16..kb+23]
__device__ __forceinline__ v16h load_afrag(const _Float16* Abase, int stride, int lane, int ktile) {
    int m  = lane & 15;
    int kb = ktile * 32 + ((lane & 16) ? 8 : 0);
    const _Float16* p = Abase + m * stride + kb;
    union { v16h v; v8h h[2]; } u;
    u.h[0] = *(const v8h*)(p);
    u.h[1] = *(const v8h*)(p + 16);
    return u.v;
}

__device__ __forceinline__ void store_f16(_Float16* D, int stride, int mtile, int ntile,
                                          v8f c, bool relu, int lane) {
    int m0 = mtile * 16 + ((lane & 16) ? 8 : 0);
    int n  = ntile * 16 + (lane & 15);
#pragma unroll
    for (int r = 0; r < 8; ++r) {
        float v = c[r];
        if (relu && v < 0.f) v = 0.f;
        D[(m0 + r) * stride + n] = (_Float16)v;
    }
}

__device__ __forceinline__ void store_f32(float* D, int stride, int mtile, int ntile,
                                          v8f c, int lane) {
    int m0 = mtile * 16 + ((lane & 16) ? 8 : 0);
    int n  = ntile * 16 + (lane & 15);
#pragma unroll
    for (int r = 0; r < 8; ++r) D[(m0 + r) * stride + n] = c[r];
}

// Single output tile (used by small post-MLP stages)
template <int KSTEPS, int NT>
__device__ __forceinline__ v8f gemm_tile(const _Float16* A, int strideA, int mtile,
                                         const _Float16* fragB, int ntile,
                                         const float* bias, int lane) {
    float bv = bias[ntile * 16 + (lane & 15)];
    v8f acc;
#pragma unroll
    for (int r = 0; r < 8; ++r) acc[r] = bv;
    const _Float16* Abase = A + mtile * 16 * strideA;
#pragma unroll
    for (int kt = 0; kt < KSTEPS; ++kt) {
        v16h a = load_afrag(Abase, strideA, lane, kt);
        v16h b = *(const v16h*)(fragB + (size_t)(kt * NT + ntile) * 512 + lane * 16);
        acc = __builtin_amdgcn_wmma_f32_16x16x32_f16(false, a, false, b, (short)0, acc, false, false);
    }
    return acc;
}

// Full layer: hoist B fragments + bias into registers, sweep M-tiles (8 waves)
template <int KSTEPS, int NT, int MTPW, bool RELU>
__device__ __forceinline__ void layer_f16(const _Float16* A, int strideA,
                                          _Float16* D, int strideD,
                                          const _Float16* fragB, const float* bias,
                                          int wave, int lane) {
#pragma unroll
    for (int nt = 0; nt < NT; ++nt) {
        v16h B[KSTEPS];
#pragma unroll
        for (int kt = 0; kt < KSTEPS; ++kt)
            B[kt] = *(const v16h*)(fragB + (size_t)(kt * NT + nt) * 512 + lane * 16);
        float bv = bias[nt * 16 + (lane & 15)];
#pragma unroll
        for (int m = 0; m < MTPW; ++m) {
            int mt = wave * MTPW + m;
            v8f acc;
#pragma unroll
            for (int r = 0; r < 8; ++r) acc[r] = bv;
            const _Float16* Ab = A + mt * 16 * strideA;
#pragma unroll
            for (int kt = 0; kt < KSTEPS; ++kt)
                acc = __builtin_amdgcn_wmma_f32_16x16x32_f16(
                    false, load_afrag(Ab, strideA, lane, kt), false, B[kt], (short)0, acc, false, false);
            store_f16(D, strideD, mt, nt, acc, RELU, lane);
        }
    }
}

template <int KSTEPS, int NT, int MTPW>
__device__ __forceinline__ void layer_f32(const _Float16* A, int strideA,
                                          float* D, int strideD,
                                          const _Float16* fragB, const float* bias,
                                          int wave, int lane) {
#pragma unroll
    for (int nt = 0; nt < NT; ++nt) {
        v16h B[KSTEPS];
#pragma unroll
        for (int kt = 0; kt < KSTEPS; ++kt)
            B[kt] = *(const v16h*)(fragB + (size_t)(kt * NT + nt) * 512 + lane * 16);
        float bv = bias[nt * 16 + (lane & 15)];
#pragma unroll
        for (int m = 0; m < MTPW; ++m) {
            int mt = wave * MTPW + m;
            v8f acc;
#pragma unroll
            for (int r = 0; r < 8; ++r) acc[r] = bv;
            const _Float16* Ab = A + mt * 16 * strideA;
#pragma unroll
            for (int kt = 0; kt < KSTEPS; ++kt)
                acc = __builtin_amdgcn_wmma_f32_16x16x32_f16(
                    false, load_afrag(Ab, strideA, lane, kt), false, B[kt], (short)0, acc, false, false);
            store_f32(D, strideD, mt, nt, acc, lane);
        }
    }
}

// ---------------- prep: fragmentize all weights into f16 B-operand layout ----------------
struct Ent { const float* src; int K; int N; int KT; int NT; };

__global__ __launch_bounds__(256) void prep_kernel(Params P, _Float16* frags) {
    Ent ents[39];
    int ne = 0;
    ents[ne++] = {P.Wenc, 16, 32, 1, 2};
    for (int c = 0; c < 3; ++c) ents[ne++] = {P.sw0 + c*32*32, 32, 32, 1, 2};
    for (int c = 0; c < 3; ++c) ents[ne++] = {P.sw1 + c*32*32, 32, 32, 1, 2};
    for (int c = 0; c < 3; ++c) ents[ne++] = {P.rw0 + c*65*64, 65, 64, 3, 4};
    for (int c = 0; c < 3; ++c) ents[ne++] = {P.rw1 + c*64*32, 64, 32, 2, 2};
    for (int c = 0; c < 3; ++c) ents[ne++] = {P.rw2 + c*32*32, 32, 32, 1, 2};
    for (int c = 0; c < 3; ++c) ents[ne++] = {P.aw0 + c*65*64, 65, 64, 3, 4};
    for (int c = 0; c < 3; ++c) ents[ne++] = {P.aw1 + c*64*32, 64, 32, 2, 2};
    for (int c = 0; c < 3; ++c) ents[ne++] = {P.fw0 + c*32*32, 32, 32, 1, 2};
    for (int c = 0; c < 3; ++c) ents[ne++] = {P.fw1 + c*32*32, 32, 32, 1, 2};
    for (int c = 0; c < 3; ++c) ents[ne++] = {P.fw2 + c*32*32, 32, 32, 1, 2};
    for (int c = 0; c < 3; ++c) ents[ne++] = {P.ow0 + c*64*32, 64, 32, 2, 2};
    for (int c = 0; c < 3; ++c) ents[ne++] = {P.ow1 + c*32*32, 32, 32, 1, 2};
    ents[ne++] = {P.Wagg1, 96, 32, 3, 2};
    ents[ne++] = {P.Wagg2, 32, 32, 1, 2};

    int g = blockIdx.x * blockDim.x + threadIdx.x;
    if (g >= NFRAGS * 512) return;
    int frag = g >> 9, rem = g & 511, lane = rem >> 4, e = rem & 15;
    int base = 0, ei = 0;
    for (; ei < ne; ++ei) {
        int fc = ents[ei].KT * ents[ei].NT;
        if (frag < base + fc) break;
        base += fc;
    }
    int lf = frag - base;
    int kt = lf / ents[ei].NT, nt = lf % ents[ei].NT;
    int K = kt * 32 + (e & 7) + ((e & 8) ? 16 : 0) + ((lane & 16) ? 8 : 0);
    int N = nt * 16 + (lane & 15);
    float v = (K < ents[ei].K && N < ents[ei].N) ? ents[ei].src[K * ents[ei].N + N] : 0.f;
    frags[g] = (_Float16)v;
}

// ---------------- encoder: s = z @ Wenc + benc ----------------
__global__ __launch_bounds__(256) void enc_kernel(Params P, const _Float16* frags, float* sFg) {
    __shared__ __align__(32) _Float16 zA[128 * 32];
    int tid = threadIdx.x, lane = tid & 31, wave = tid >> 5;
    int rbase = blockIdx.x * 128;
    for (int idx = tid; idx < 128 * 32; idx += 256) {
        int row = idx >> 5, col = idx & 31;
        float v = (col < 16) ? P.z[(size_t)(rbase + row) * 16 + col] : 0.f;
        zA[idx] = (_Float16)v;
    }
    __syncthreads();
    int mt = wave;  // 8 waves, 8 M-tiles of 16 rows
#pragma unroll
    for (int nt = 0; nt < 2; ++nt) {
        v8f acc = gemm_tile<1, 2>(zA, 32, mt, frags + (size_t)FB_WENC * 512, nt, P.benc, lane);
        int m0 = mt * 16 + ((lane & 16) ? 8 : 0);
        int n  = nt * 16 + (lane & 15);
#pragma unroll
        for (int r = 0; r < 8; ++r) sFg[(size_t)(rbase + m0 + r) * 32 + n] = acc[r];
    }
}

// ---------------- fused interaction-network core: 4 samples / block ----------------
__global__ __launch_bounds__(256) void core_kernel(Params P, const float* sFg,
                                                   const _Float16* frags, float* coreOut) {
    const int c = blockIdx.y;          // which core (0..2)
    const int sbase = blockIdx.x * 4;  // 4 samples per block
    const int tid = threadIdx.x, lane = tid & 31, wave = tid >> 5;

    __shared__ __align__(1024) char smem[SM_SIZE];
    _Float16* comb  = (_Float16*)(smem + SM_COMB);   // 256 x 96
    float*    rel2F = (float*)   (smem + SM_REL2F);  // 256 x 32 (aliases comb)
    _Float16* h0    = (_Float16*)(smem + SM_H0);     // 256 x 64
    _Float16* h1    = (_Float16*)(smem + SM_H1);     // 256 x 32
    _Float16* catb  = (_Float16*)(smem + SM_CATB);   // 32 x 64 (aliases h1)
    _Float16* pA    = (_Float16*)(smem + SM_PA);     // 32 x 32 (aliases h1)
    _Float16* pB    = (_Float16*)(smem + SM_PB);     // 32 x 32 (aliases h1)
    _Float16* sH    = (_Float16*)(smem + SM_SH);     // 32 x 32
    float*    sF    = (float*)   (smem + SM_SF);     // 32 x 32
    float*    att2F = (float*)   (smem + SM_ATT2F);  // 256
    float*    relsd = (float*)   (smem + SM_RELSD);  // 32 x 32

    // 1. load s block (32 object-rows x 32)
    for (int idx = tid; idx < 32 * 32; idx += 256) {
        int row = idx >> 5, col = idx & 31;
        float v = sFg[(size_t)(sbase * 8 + row) * 32 + col];
        sF[idx] = v; sH[idx] = (_Float16)v;
    }
    __syncthreads();

    // 2. build comb: 256 pair rows x [s_i(32) | s_j(32) | dist | pad->96]
    {
        int r = tid;  // one pair-row per thread
        int sl = r >> 6, i = (r >> 3) & 7, j = r & 7;
        const float* si = sF + (sl * 8 + i) * 32;
        const float* sj = sF + (sl * 8 + j) * 32;
        float acc = 0.f;
#pragma unroll
        for (int k = 0; k < 32; ++k) { float d = si[k] - sj[k]; acc += d * d; }
        comb[r * 96 + 64] = (_Float16)sqrtf(acc + 1e-12f);
#pragma unroll
        for (int k = 65; k < 96; ++k) comb[r * 96 + k] = (_Float16)0.f;
    }
    for (int idx = tid; idx < 256 * 64; idx += 256) {
        int r = idx >> 6, cc = idx & 63;
        int sl = r >> 6, i = (r >> 3) & 7, j = r & 7;
        comb[r * 96 + cc] = (cc < 32) ? sH[(sl * 8 + i) * 32 + cc]
                                      : sH[(sl * 8 + j) * 32 + (cc - 32)];
    }
    __syncthreads();

    // ---- attention path first (so rel2F can alias comb later) ----
    // att0: h0 = relu(comb @ aw0 + ab0)
    layer_f16<3, 4, 2, true>(comb, 96, h0, 64,
                             frags + (size_t)(FB_AW0 + c * 12) * 512, P.ab0 + c * 64, wave, lane);
    __syncthreads();
    // att1: h1 = relu(h0 @ aw1 + ab1)
    layer_f16<2, 2, 2, true>(h0, 64, h1, 32,
                             frags + (size_t)(FB_AW1 + c * 4) * 512, P.ab1 + c * 32, wave, lane);
    __syncthreads();
    // att2 = sigmoid(h1 @ aw2 + ab2)  (32->1, VALU)
    {
        int r = tid;
        const float* aw2 = P.aw2 + c * 32;
        float x = P.ab2[c];
#pragma unroll
        for (int k = 0; k < 32; ++k) x += (float)h1[r * 32 + k] * aw2[k];
        att2F[r] = 1.f / (1.f + __expf(-x));
    }
    __syncthreads();

    // ---- relational path ----
    // rel0: h0 = relu(comb @ rw0 + rb0)   (last reader of comb)
    layer_f16<3, 4, 2, true>(comb, 96, h0, 64,
                             frags + (size_t)(FB_RW0 + c * 12) * 512, P.rb0 + c * 64, wave, lane);
    __syncthreads();
    // rel1: h1 = relu(h0 @ rw1 + rb1)
    layer_f16<2, 2, 2, true>(h0, 64, h1, 32,
                             frags + (size_t)(FB_RW1 + c * 4) * 512, P.rb1 + c * 32, wave, lane);
    __syncthreads();
    // rel2F = h1 @ rw2 + rb2  (f32, overwrites comb region)
    layer_f32<1, 2, 2>(h1, 32, rel2F, 32,
                       frags + (size_t)(FB_RW2 + c * 2) * 512, P.rb2 + c * 32, wave, lane);
    __syncthreads();

    // rel_sd: masked attention-weighted aggregation over partners
    for (int idx = tid; idx < 1024; idx += 256) {
        int row = idx >> 5, cl = idx & 31;
        int sl = row >> 3, i = row & 7;
        float acc = 0.f;
#pragma unroll
        for (int j = 0; j < 8; ++j) {
            if (j == i) continue;
            int pr = sl * 64 + i * 8 + j;
            acc += rel2F[pr * 32 + cl] * att2F[pr];
        }
        relsd[idx] = acc;
    }
    __syncthreads();

    // ---- post MLPs: M=32 => 2Mt x 2Nt = 4 tiles on waves 0..3 ----
    const int mt = wave >> 1, nt = wave & 1;
    // a. pA = relu(s @ sw0 + sb0)
    if (wave < 4) {
        v8f acc = gemm_tile<1, 2>(sH, 32, mt, frags + (size_t)(FB_SW0 + c * 2) * 512, nt, P.sb0 + c * 32, lane);
        store_f16(pA, 32, mt, nt, acc, true, lane);
    }
    __syncthreads();
    // b. dyn = (pA @ sw1 + sb1) + rel_sd  -> pB (f16)
    if (wave < 4) {
        v8f acc = gemm_tile<1, 2>(pA, 32, mt, frags + (size_t)(FB_SW1 + c * 2) * 512, nt, P.sb1 + c * 32, lane);
        int m0 = mt * 16 + ((lane & 16) ? 8 : 0), n = nt * 16 + (lane & 15);
#pragma unroll
        for (int r = 0; r < 8; ++r)
            pB[(m0 + r) * 32 + n] = (_Float16)(acc[r] + relsd[(m0 + r) * 32 + n]);
    }
    __syncthreads();
    // c. pA = relu(pB @ fw0 + fb0)
    if (wave < 4) {
        v8f acc = gemm_tile<1, 2>(pB, 32, mt, frags + (size_t)(FB_FW0 + c * 2) * 512, nt, P.fb0 + c * 32, lane);
        store_f16(pA, 32, mt, nt, acc, true, lane);
    }
    __syncthreads();
    // d. pB = relu(pA @ fw1 + fb1)
    if (wave < 4) {
        v8f acc = gemm_tile<1, 2>(pA, 32, mt, frags + (size_t)(FB_FW1 + c * 2) * 512, nt, P.fb1 + c * 32, lane);
        store_f16(pB, 32, mt, nt, acc, true, lane);
    }
    __syncthreads();
    // e. pA = pB @ fw2 + fb2 (no relu) -> affected
    if (wave < 4) {
        v8f acc = gemm_tile<1, 2>(pB, 32, mt, frags + (size_t)(FB_FW2 + c * 2) * 512, nt, P.fb2 + c * 32, lane);
        store_f16(pA, 32, mt, nt, acc, false, lane);
    }
    __syncthreads();
    // f. cat = [aff | s]
    for (int idx = tid; idx < 32 * 64; idx += 256) {
        int row = idx >> 6, cc = idx & 63;
        catb[idx] = (cc < 32) ? pA[row * 32 + cc] : sH[row * 32 + (cc - 32)];
    }
    __syncthreads();
    // g. pB = relu(cat @ ow0 + ob0)  (K=64)
    if (wave < 4) {
        v8f acc = gemm_tile<2, 2>(catb, 64, mt, frags + (size_t)(FB_OW0 + c * 4) * 512, nt, P.ob0 + c * 32, lane);
        store_f16(pB, 32, mt, nt, acc, true, lane);
    }
    __syncthreads();
    // h. coreOut = pB @ ow1 + ob1 (f32 -> global)
    if (wave < 4) {
        v8f acc = gemm_tile<1, 2>(pB, 32, mt, frags + (size_t)(FB_OW1 + c * 2) * 512, nt, P.ob1 + c * 32, lane);
        float* dst = coreOut + (size_t)c * NROWS * CLS;
        int m0 = mt * 16 + ((lane & 16) ? 8 : 0), n = nt * 16 + (lane & 15);
#pragma unroll
        for (int r = 0; r < 8; ++r)
            dst[(size_t)(sbase * 8 + m0 + r) * 32 + n] = acc[r];
    }
}

// ---------------- final aggregation: relu(cat3 @ Wagg1 + b) @ Wagg2 + b ----------------
__global__ __launch_bounds__(128) void agg_kernel(Params P, const _Float16* frags,
                                                  const float* coreOut, float* out) {
    __shared__ __align__(32) _Float16 A[64 * 96];
    __shared__ __align__(32) _Float16 H[64 * 32];
    int tid = threadIdx.x, lane = tid & 31, wave = tid >> 5;
    int rbase = blockIdx.x * 64;
    for (int idx = tid; idx < 64 * 96; idx += 128) {
        int row = idx / 96, col = idx - row * 96;
        int core = col >> 5, cc = col & 31;
        A[idx] = (_Float16)coreOut[(size_t)core * NROWS * CLS + (size_t)(rbase + row) * 32 + cc];
    }
    __syncthreads();
#pragma unroll
    for (int t = 0; t < 2; ++t) {  // 4Mt x 2Nt = 8 tiles over 4 waves
        int tile = wave * 2 + t, mt = tile >> 1, nt = tile & 1;
        v8f acc = gemm_tile<3, 2>(A, 96, mt, frags + (size_t)FB_WAGG1 * 512, nt, P.bagg1, lane);
        store_f16(H, 32, mt, nt, acc, true, lane);
    }
    __syncthreads();
#pragma unroll
    for (int t = 0; t < 2; ++t) {
        int tile = wave * 2 + t, mt = tile >> 1, nt = tile & 1;
        v8f acc = gemm_tile<1, 2>(H, 32, mt, frags + (size_t)FB_WAGG2 * 512, nt, P.bagg2, lane);
        int m0 = mt * 16 + ((lane & 16) ? 8 : 0), n = nt * 16 + (lane & 15);
#pragma unroll
        for (int r = 0; r < 8; ++r)
            out[(size_t)(rbase + m0 + r) * 32 + n] = acc[r];
    }
}

extern "C" void kernel_launch(void* const* d_in, const int* in_sizes, int n_in,
                              void* d_out, int out_size, void* d_ws, size_t ws_size,
                              hipStream_t stream) {
    Params P;
    P.z     = (const float*)d_in[0];  P.Wenc  = (const float*)d_in[1];  P.benc  = (const float*)d_in[2];
    P.sw0   = (const float*)d_in[3];  P.sb0   = (const float*)d_in[4];
    P.sw1   = (const float*)d_in[5];  P.sb1   = (const float*)d_in[6];
    P.rw0   = (const float*)d_in[7];  P.rb0   = (const float*)d_in[8];
    P.rw1   = (const float*)d_in[9];  P.rb1   = (const float*)d_in[10];
    P.rw2   = (const float*)d_in[11]; P.rb2   = (const float*)d_in[12];
    P.aw0   = (const float*)d_in[13]; P.ab0   = (const float*)d_in[14];
    P.aw1   = (const float*)d_in[15]; P.ab1   = (const float*)d_in[16];
    P.aw2   = (const float*)d_in[17]; P.ab2   = (const float*)d_in[18];
    P.fw0   = (const float*)d_in[19]; P.fb0   = (const float*)d_in[20];
    P.fw1   = (const float*)d_in[21]; P.fb1   = (const float*)d_in[22];
    P.fw2   = (const float*)d_in[23]; P.fb2   = (const float*)d_in[24];
    P.ow0   = (const float*)d_in[25]; P.ob0   = (const float*)d_in[26];
    P.ow1   = (const float*)d_in[27]; P.ob1   = (const float*)d_in[28];
    P.Wagg1 = (const float*)d_in[29]; P.bagg1 = (const float*)d_in[30];
    P.Wagg2 = (const float*)d_in[31]; P.bagg2 = (const float*)d_in[32];

    char* ws = (char*)d_ws;
    _Float16* frags = (_Float16*)(ws + WS_FRAGS);
    float*    sFg   = (float*)(ws + WS_SF);
    float*    cOut  = (float*)(ws + WS_CORE);

    prep_kernel<<<(NFRAGS * 512 + 255) / 256, 256, 0, stream>>>(P, frags);
    enc_kernel<<<NROWS / 128, 256, 0, stream>>>(P, frags, sFg);
    dim3 grid(NSAMP / 4, 3);
    core_kernel<<<grid, 256, 0, stream>>>(P, sFg, frags, cOut);
    agg_kernel<<<NROWS / 64, 128, 0, stream>>>(P, frags, cOut, (float*)d_out);
}
